// SparseResidualBlock_16758962389329
// MI455X (gfx1250) — matrix operand order repaired and checked by reading, not compiled, not run on hardware
//
#include <hip/hip_runtime.h>

typedef __bf16 bf16;
typedef __attribute__((ext_vector_type(8)))  __bf16 v8bf;
typedef __attribute__((ext_vector_type(16))) __bf16 v16bf;
typedef __attribute__((ext_vector_type(8)))  float  v8f;

#define NPTS   300000
#define NTAPS  27
#define NSUP   9375    // NPTS / 32 (exact): 32-row super-tiles, one per wave

__device__ __forceinline__ v16bf cat8(v8bf lo, v8bf hi) {
  v16bf a;
#pragma unroll
  for (int i = 0; i < 8; ++i) { a[i] = lo[i]; a[i + 8] = hi[i]; }
  return a;
}

__device__ __forceinline__ float leaky(float x) { return x >= 0.f ? x : 0.2f * x; }

// Branch-free gather remap: invalid neighbor -> zero-row sentinel at NPTS.
__device__ __forceinline__ int remap(int idx) { return idx < 0 ? NPTS : idx; }

// One 16x32 bf16 A fragment: lane (r=l&15, khalf=l>>4) holds cols
// [khalf*8,+8) and [khalf*8+16,+8) of its gathered row (WMMA A layout).
__device__ __forceinline__ v16bf loadA32(const bf16* __restrict__ src,
                                         int idx, int khalf) {
  const v8bf* p = (const v8bf*)(src + (long)idx * 32 + khalf * 8);
  return cat8(p[0], p[2]);
}

// ---------------------------------------------------------------------------
// Pack W1 / W2 / Wd (fp32, row-major [cin][cout]) into bf16 WMMA B-operand
// layout: per (tap, col-tile, k-step) a 32-lane x 16-element block where
// lane l = (n = l&15, khalf = l>>4) holds K = khalf*8 + (j&7) + (j>>3)*16  (j = 0..15).
// pW1: 27*4*512 = 55296   pW2: 27*4*2*512 = 110592   pWd: 4*512 = 2048
// ---------------------------------------------------------------------------
__global__ void pack_weights_kernel(const float* __restrict__ W1,
                                    const float* __restrict__ W2,
                                    const float* __restrict__ Wd,
                                    bf16* __restrict__ pW1,
                                    bf16* __restrict__ pW2,
                                    bf16* __restrict__ pWd) {
  int e = blockIdx.x * blockDim.x + threadIdx.x;
  if (e < 55296) {                         // W1: [k][ct][lane][j]
    int j = e & 15, lane = (e >> 4) & 31, ct = (e >> 9) & 3, k = e >> 11;
    int n   = ct * 16 + (lane & 15);
    int cin = (lane >> 4) * 8 + (j & 7) + (j >> 3) * 16;
    pW1[e] = (bf16)W1[(k * 32 + cin) * 64 + n];
  } else if (e < 165888) {                 // W2: [k][ct][ks][lane][j]
    int e2 = e - 55296;
    int j = e2 & 15, lane = (e2 >> 4) & 31, ks = (e2 >> 9) & 1,
        ct = (e2 >> 10) & 3, k = e2 >> 12;
    int n   = ct * 16 + (lane & 15);
    int cin = ks * 32 + (lane >> 4) * 8 + (j & 7) + (j >> 3) * 16;
    pW2[e2] = (bf16)W2[(k * 64 + cin) * 64 + n];
  } else if (e < 167936) {                 // Wd: [ct][lane][j]
    int e3 = e - 165888;
    int j = e3 & 15, lane = (e3 >> 4) & 31, ct = (e3 >> 9) & 3;
    int n   = ct * 16 + (lane & 15);
    int cin = (lane >> 4) * 8 + (j & 7) + (j >> 3) * 16;
    pWd[e3] = (bf16)Wd[cin * 64 + n];
  }
}

// Convert features to bf16, write the featBf zero row (index NPTS) and the
// hBf zero row (index NPTS). Total threads: (NPTS+1)*32 + 64.
__global__ void cvt_feats_kernel(const float* __restrict__ f,
                                 bf16* __restrict__ fb,
                                 bf16* __restrict__ hzero) {
  int i = blockIdx.x * blockDim.x + threadIdx.x;
  if (i < NPTS * 32) {
    fb[i] = (bf16)f[i];
  } else if (i < (NPTS + 1) * 32) {
    fb[i] = (bf16)0.f;                   // feature zero row
  } else if (i < (NPTS + 1) * 32 + 64) {
    hzero[i - (NPTS + 1) * 32] = (bf16)0.f;   // h zero row
  }
}

// ---------------------------------------------------------------------------
// Layer 1: h = leaky( sum_k gather(features)[32x32] @ W1[k][32x64] )
// One wave owns a 32-row super-tile = two 16-row WMMA tiles sharing B.
// Branch-free gathers + one-tap-ahead prefetch, unrolled x2 so the software
// pipeline rotation is pure register ping-pong (no mov / hazard nops).
// ---------------------------------------------------------------------------
__global__ __launch_bounds__(256) void conv1_kernel(const bf16* __restrict__ featBf,
                                                    const int*  __restrict__ nidx,
                                                    const bf16* __restrict__ pW1,
                                                    bf16* __restrict__ hBf) {
  const int lane = threadIdx.x & 31;
  const int sup  = blockIdx.x * 8 + (threadIdx.x >> 5);
  if (sup >= NSUP) return;               // wave-uniform: EXEC stays all-ones
  const int r     = lane & 15;
  const int khalf = lane >> 4;
  const int base  = sup * 32;
  const int* np0  = nidx + (long)(base + r) * NTAPS;
  const int* np1  = nidx + (long)(base + 16 + r) * NTAPS;

  v8f acc0[4] = {};
  v8f acc1[4] = {};

  v16bf a0 = loadA32(featBf, remap(np0[0]), khalf);
  v16bf a1 = loadA32(featBf, remap(np1[0]), khalf);

#pragma unroll 2
  for (int k = 0; k < NTAPS - 1; ++k) {
    const int ni0 = remap(np0[k + 1]);
    const int ni1 = remap(np1[k + 1]);

    const bf16* wb = pW1 + k * 2048 + lane * 16;
    v16bf b[4];
#pragma unroll
    for (int ct = 0; ct < 4; ++ct) b[ct] = *(const v16bf*)(wb + ct * 512);

    v16bf na0 = loadA32(featBf, ni0, khalf);
    v16bf na1 = loadA32(featBf, ni1, khalf);

#pragma unroll
    for (int ct = 0; ct < 4; ++ct) {
      acc0[ct] = __builtin_amdgcn_wmma_f32_16x16x32_bf16(
          false, a0, false, b[ct], (short)0, acc0[ct], false, false);
      acc1[ct] = __builtin_amdgcn_wmma_f32_16x16x32_bf16(
          false, a1, false, b[ct], (short)0, acc1[ct], false, false);
    }
    a0 = na0; a1 = na1;
  }
  {                                      // peeled last tap
    const bf16* wb = pW1 + (NTAPS - 1) * 2048 + lane * 16;
    v16bf b[4];
#pragma unroll
    for (int ct = 0; ct < 4; ++ct) b[ct] = *(const v16bf*)(wb + ct * 512);
#pragma unroll
    for (int ct = 0; ct < 4; ++ct) {
      acc0[ct] = __builtin_amdgcn_wmma_f32_16x16x32_bf16(
          false, a0, false, b[ct], (short)0, acc0[ct], false, false);
      acc1[ct] = __builtin_amdgcn_wmma_f32_16x16x32_bf16(
          false, a1, false, b[ct], (short)0, acc1[ct], false, false);
    }
  }

  // C/D layout: lane half khalf -> rows v + khalf*8; col = ct*16 + r
#pragma unroll
  for (int ct = 0; ct < 4; ++ct) {
#pragma unroll
    for (int v = 0; v < 8; ++v) {
      int row0 = base + v + khalf * 8;
      int row1 = base + 16 + v + khalf * 8;
      hBf[row0 * 64 + ct * 16 + r] = (bf16)leaky(acc0[ct][v]);
      hBf[row1 * 64 + ct * 16 + r] = (bf16)leaky(acc1[ct][v]);
    }
  }
}

// ---------------------------------------------------------------------------
// Layer 2 + skip: out = leaky( sum_k gather(h)[32x64] @ W2[k] + feat @ Wd )
// K=64 -> two WMMA k-steps; per tap: 8 gather loads + 16 B loads -> 16 WMMAs.
// ---------------------------------------------------------------------------
__global__ __launch_bounds__(256) void conv2_kernel(const bf16* __restrict__ featBf,
                                                    const bf16* __restrict__ hBf,
                                                    const int*  __restrict__ nidx,
                                                    const bf16* __restrict__ pW2,
                                                    const bf16* __restrict__ pWd,
                                                    float* __restrict__ out) {
  const int lane = threadIdx.x & 31;
  const int sup  = blockIdx.x * 8 + (threadIdx.x >> 5);
  if (sup >= NSUP) return;
  const int r     = lane & 15;
  const int khalf = lane >> 4;
  const int base  = sup * 32;
  const int* np0  = nidx + (long)(base + r) * NTAPS;
  const int* np1  = nidx + (long)(base + 16 + r) * NTAPS;

  v8f acc0[4], acc1[4];
  {                                      // identity path: featBf[row,:] @ Wd
    v16bf a0 = loadA32(featBf, base + r, khalf);
    v16bf a1 = loadA32(featBf, base + 16 + r, khalf);
#pragma unroll
    for (int ct = 0; ct < 4; ++ct) {
      v16bf b = *(const v16bf*)(pWd + ct * 512 + lane * 16);
      v8f z = {};
      acc0[ct] = __builtin_amdgcn_wmma_f32_16x16x32_bf16(
          false, a0, false, b, (short)0, z, false, false);
      acc1[ct] = __builtin_amdgcn_wmma_f32_16x16x32_bf16(
          false, a1, false, b, (short)0, z, false, false);
    }
  }

  // prime tap 0 gathers (rows of h, 64 wide: two k-step fragments each)
  int i0 = remap(np0[0]);
  int i1 = remap(np1[0]);
  const v8bf* q0 = (const v8bf*)(hBf + (long)i0 * 64 + khalf * 8);
  const v8bf* q1 = (const v8bf*)(hBf + (long)i1 * 64 + khalf * 8);
  v16bf a0k0 = cat8(q0[0], q0[2]), a0k1 = cat8(q0[4], q0[6]);
  v16bf a1k0 = cat8(q1[0], q1[2]), a1k1 = cat8(q1[4], q1[6]);

#pragma unroll 2
  for (int k = 0; k < NTAPS - 1; ++k) {
    const int ni0 = remap(np0[k + 1]);
    const int ni1 = remap(np1[k + 1]);

    const bf16* wb = pW2 + k * 4096 + lane * 16;
    v16bf b[8];                          // [ct*2 + ks]
#pragma unroll
    for (int i = 0; i < 8; ++i) b[i] = *(const v16bf*)(wb + i * 512);

    const v8bf* nq0 = (const v8bf*)(hBf + (long)ni0 * 64 + khalf * 8);
    const v8bf* nq1 = (const v8bf*)(hBf + (long)ni1 * 64 + khalf * 8);
    v16bf na0k0 = cat8(nq0[0], nq0[2]), na0k1 = cat8(nq0[4], nq0[6]);
    v16bf na1k0 = cat8(nq1[0], nq1[2]), na1k1 = cat8(nq1[4], nq1[6]);

#pragma unroll
    for (int ct = 0; ct < 4; ++ct) {
      acc0[ct] = __builtin_amdgcn_wmma_f32_16x16x32_bf16(
          false, a0k0, false, b[ct * 2 + 0], (short)0, acc0[ct], false, false);
      acc0[ct] = __builtin_amdgcn_wmma_f32_16x16x32_bf16(
          false, a0k1, false, b[ct * 2 + 1], (short)0, acc0[ct], false, false);
      acc1[ct] = __builtin_amdgcn_wmma_f32_16x16x32_bf16(
          false, a1k0, false, b[ct * 2 + 0], (short)0, acc1[ct], false, false);
      acc1[ct] = __builtin_amdgcn_wmma_f32_16x16x32_bf16(
          false, a1k1, false, b[ct * 2 + 1], (short)0, acc1[ct], false, false);
    }
    a0k0 = na0k0; a0k1 = na0k1; a1k0 = na1k0; a1k1 = na1k1;
  }
  {                                      // peeled last tap
    const bf16* wb = pW2 + (NTAPS - 1) * 4096 + lane * 16;
    v16bf b[8];
#pragma unroll
    for (int i = 0; i < 8; ++i) b[i] = *(const v16bf*)(wb + i * 512);
#pragma unroll
    for (int ct = 0; ct < 4; ++ct) {
      acc0[ct] = __builtin_amdgcn_wmma_f32_16x16x32_bf16(
          false, a0k0, false, b[ct * 2 + 0], (short)0, acc0[ct], false, false);
      acc0[ct] = __builtin_amdgcn_wmma_f32_16x16x32_bf16(
          false, a0k1, false, b[ct * 2 + 1], (short)0, acc0[ct], false, false);
      acc1[ct] = __builtin_amdgcn_wmma_f32_16x16x32_bf16(
          false, a1k0, false, b[ct * 2 + 0], (short)0, acc1[ct], false, false);
      acc1[ct] = __builtin_amdgcn_wmma_f32_16x16x32_bf16(
          false, a1k1, false, b[ct * 2 + 1], (short)0, acc1[ct], false, false);
    }
  }

#pragma unroll
  for (int ct = 0; ct < 4; ++ct) {
#pragma unroll
    for (int v = 0; v < 8; ++v) {
      int row0 = base + v + khalf * 8;
      int row1 = base + 16 + v + khalf * 8;
      out[row0 * 64 + ct * 16 + r] = leaky(acc0[ct][v]);
      out[row1 * 64 + ct * 16 + r] = leaky(acc1[ct][v]);
    }
  }
}

extern "C" void kernel_launch(void* const* d_in, const int* in_sizes, int n_in,
                              void* d_out, int out_size, void* d_ws, size_t ws_size,
                              hipStream_t stream) {
  const float* features = (const float*)d_in[0];
  const int*   nidx     = (const int*)  d_in[1];
  const float* W1       = (const float*)d_in[2];
  const float* W2       = (const float*)d_in[3];
  const float* Wd       = (const float*)d_in[4];
  float* out = (float*)d_out;

  // Workspace layout (bf16 elements), all chunks 32B-aligned. featBf and hBf
  // each carry one extra all-zero row (index NPTS) as the gather sentinel.
  bf16* pW1    = (bf16*)d_ws;                       //  55296
  bf16* pW2    = pW1 + 55296;                       // 110592
  bf16* pWd    = pW2 + 110592;                      //   2048
  bf16* featBf = pWd + 2048;                        // (NPTS+1)*32
  bf16* hBf    = featBf + (size_t)(NPTS + 1) * 32;  // (NPTS+1)*64  (~55 MB total)

  pack_weights_kernel<<<656, 256, 0, stream>>>(W1, W2, Wd, pW1, pW2, pWd);
  int cvtThreads = (NPTS + 1) * 32 + 64;
  cvt_feats_kernel<<<(cvtThreads + 255) / 256, 256, 0, stream>>>(
      features, featBf, hBf + (size_t)NPTS * 64);
  conv1_kernel<<<(NSUP + 7) / 8, 256, 0, stream>>>(featBf, nidx, pW1, hBf);
  conv2_kernel<<<(NSUP + 7) / 8, 256, 0, stream>>>(featBf, hBf, nidx, pW2, pWd, out);
}
